// NDDIso_8229157339271
// MI455X (gfx1250) — matrix operand order repaired and checked by reading, not compiled, not run on hardware
//
#include <hip/hip_runtime.h>

typedef __attribute__((ext_vector_type(16))) _Float16 v16h;
typedef __attribute__((ext_vector_type(8)))  _Float16 v8h;
typedef __attribute__((ext_vector_type(8)))  float    v8f;

#define B_TOT 512
#define T_LEN 1024
#define C_CH  64
#define H_DIM 128
#define G4H   512   // 4*H
#define BC    16    // batch rows per workgroup
#define XT    64    // x staging tile (timesteps)

__device__ __forceinline__ float fast_tanh(float x) {
#if __has_builtin(__builtin_amdgcn_tanhf)
    return __builtin_amdgcn_tanhf(x);   // v_tanh_f32 on gfx1250
#else
    return 1.0f - 2.0f / (__expf(2.0f * x) + 1.0f);
#endif
}

__global__ __launch_bounds__(256, 1)
void lstm_chan_kernel(const float* __restrict__ x,
                      const float* __restrict__ W_ih,
                      const float* __restrict__ W_hh,
                      const float* __restrict__ b_ih,
                      const float* __restrict__ b_hh,
                      const float* __restrict__ W_fc,
                      const float* __restrict__ b_fc,
                      float* __restrict__ out)
{
    // B-matrix layout [n][k] f16 (128 KB): row n contiguous in K -> 32B B-frag loads
    __shared__ __align__(32) _Float16 w_lds[G4H * H_DIM];
    __shared__ __align__(32) _Float16 h_lds[BC * H_DIM];   // h state [b][k], 4 KB
    __shared__ __align__(16) float    x_lds[XT * BC];      // 4 KB
    __shared__ float wib_lds[G4H];
    __shared__ float bias_lds[G4H];
    __shared__ float wfc_lds[H_DIM];

    const int chunk = blockIdx.x;      // 0..31  (batch chunk)
    const int c     = blockIdx.y;      // 0..63  (channel)
    const int b0    = chunk * BC;
    const int tid   = threadIdx.x;
    const int lane  = tid & 31;
    const int wave  = tid >> 5;        // 0..7
    const int hi    = lane >> 4;       // half-wave select
    const int li    = lane & 15;

    // ---- stage channel weights into LDS ----
    {
        const float4* src = (const float4*)(W_hh + (size_t)c * G4H * H_DIM);
        for (int i = tid; i < (G4H * H_DIM) / 4; i += 256) {
            float4 v = src[i];
            _Float16* d = w_lds + i * 4;
            d[0] = (_Float16)v.x; d[1] = (_Float16)v.y;
            d[2] = (_Float16)v.z; d[3] = (_Float16)v.w;
        }
        for (int i = tid; i < G4H; i += 256) {
            wib_lds[i]  = W_ih[(size_t)c * G4H + i];
            bias_lds[i] = b_ih[(size_t)c * G4H + i] + b_hh[(size_t)c * G4H + i];
        }
        for (int i = tid; i < H_DIM; i += 256) wfc_lds[i] = W_fc[(size_t)c * H_DIM + i];
        for (int i = tid; i < BC * H_DIM; i += 256) h_lds[i] = (_Float16)0.0f;
    }
    __syncthreads();

    // Wave w owns N-tiles {w, w+8, w+16, w+24}: same hidden column hcol for all 4 gates
    const int hcol = 16 * wave + li;
    float wib[4], bia[4], wibh[4], biah[4];
#pragma unroll
    for (int g = 0; g < 4; ++g) {
        wib[g]  = wib_lds[g * H_DIM + hcol];
        bia[g]  = bias_lds[g * H_DIM + hcol];
        wibh[g] = 0.5f * wib[g];      // folded sigmoid scale
        biah[g] = 0.5f * bia[g];
    }

    // ---- preload all B fragments (W_hh) into registers: loop-invariant ----
    v16h bfrag[4][4];                 // [kt][gate]
#pragma unroll
    for (int kt = 0; kt < 4; ++kt) {
        const int kb = kt * 32 + hi * 16;
#pragma unroll
        for (int g = 0; g < 4; ++g) {
            const int ntile = wave + 8 * g;
            bfrag[kt][g] = *(const v16h*)(w_lds + (ntile * 16 + li) * H_DIM + kb);
        }
    }

    const v8f vzero = {};
    v8f   acc[4];                     // per-gate accumulators (D layout)
    float cst[8];                     // cell state, D layout rows
#pragma unroll
    for (int r = 0; r < 8; ++r) cst[r] = 0.0f;

    const float* xc = x + (size_t)b0 * T_LEN * C_CH + c;

    for (int t = 0; t < T_LEN; ++t) {
        // ---- stage x tile every XT steps ----
        if ((t & (XT - 1)) == 0) {
            for (int i = tid; i < XT * BC; i += 256) {
                int tt = i >> 4;              // timestep within tile
                int bl = i & (BC - 1);        // local batch row
                x_lds[tt * BC + bl] = xc[((size_t)bl * T_LEN + (t + tt)) * C_CH];
            }
            __syncthreads();
        }

        // ---- gates = h @ W_hh^T  (reads h_lds; B frags already in VGPRs) ----
#pragma unroll
        for (int g = 0; g < 4; ++g) acc[g] = vzero;

#pragma unroll
        for (int kt = 0; kt < 4; ++kt) {
            const int ka = kt * 32 + hi * 16; // A frag: K = ka..ka+15
            const _Float16* ap = h_lds + li * H_DIM + ka;
            v8h alo = *(const v8h*)ap;
            v8h ahi = *(const v8h*)(ap + 8);
            v16h a = __builtin_shufflevector(alo, ahi,
                      0,1,2,3,4,5,6,7,8,9,10,11,12,13,14,15);
#pragma unroll
            for (int g = 0; g < 4; ++g)
                acc[g] = __builtin_amdgcn_wmma_f32_16x16x32_f16(
                    false, a, false, bfrag[kt][g], (short)0, acc[g], false, false);
        }
        __syncthreads();   // all waves done reading h_lds

        // ---- cell update + write h (D layout: row = 8*hi + r, col = hcol) ----
        {
            const int tt = t & (XT - 1);
            const float4* xp = (const float4*)(x_lds + tt * BC + hi * 8);
            float4 x0 = xp[0], x1 = xp[1];
            float xr[8] = {x0.x, x0.y, x0.z, x0.w, x1.x, x1.y, x1.z, x1.w};
            const int rowbase = hi * 8;
#pragma unroll
            for (int r = 0; r < 8; ++r) {
                float xv = xr[r];
                // sigmoid(z) = 0.5*tanh(0.5z)+0.5 with 0.5 folded into wibh/biah
                float iv = 0.5f * fast_tanh(0.5f * acc[0][r] + (xv * wibh[0] + biah[0])) + 0.5f;
                float fv = 0.5f * fast_tanh(0.5f * acc[1][r] + (xv * wibh[1] + biah[1])) + 0.5f;
                float gv = fast_tanh(acc[2][r] + (xv * wib[2] + bia[2]));
                float ov = 0.5f * fast_tanh(0.5f * acc[3][r] + (xv * wibh[3] + biah[3])) + 0.5f;
                float cn = fv * cst[r] + iv * gv;
                cst[r] = cn;
                float hn = ov * fast_tanh(cn);
                h_lds[(rowbase + r) * H_DIM + hcol] = (_Float16)hn;
            }
        }
        __syncthreads();   // h_lds writes visible for next step
    }

    // ---- out[b, c] = h_last . W_fc[c] + b_fc[c] ----
    if (tid < BC) {
        const int b = tid;
        float s = 0.0f;
#pragma unroll 8
        for (int k = 0; k < H_DIM; ++k)
            s += (float)h_lds[b * H_DIM + k] * wfc_lds[k];
        out[(size_t)(b0 + b) * C_CH + c] = s + b_fc[c];
    }
}

extern "C" void kernel_launch(void* const* d_in, const int* in_sizes, int n_in,
                              void* d_out, int out_size, void* d_ws, size_t ws_size,
                              hipStream_t stream) {
    (void)in_sizes; (void)n_in; (void)out_size; (void)d_ws; (void)ws_size;
    const float* x    = (const float*)d_in[0];
    const float* W_ih = (const float*)d_in[1];
    const float* W_hh = (const float*)d_in[2];
    const float* b_ih = (const float*)d_in[3];
    const float* b_hh = (const float*)d_in[4];
    const float* W_fc = (const float*)d_in[5];
    const float* b_fc = (const float*)d_in[6];
    float* out = (float*)d_out;

    dim3 grid(B_TOT / BC, C_CH);
    lstm_chan_kernel<<<grid, dim3(256), 0, stream>>>(
        x, W_ih, W_hh, b_ih, b_hh, W_fc, b_fc, out);
}
